// TokenResampler_56994216018127
// MI455X (gfx1250) — matrix operand order repaired and checked by reading, not compiled, not run on hardware
//
#include <hip/hip_runtime.h>
#include <hip/hip_bf16.h>

typedef __attribute__((ext_vector_type(16))) _Float16 v16h;
typedef __attribute__((ext_vector_type(8)))  _Float16 v8h;
typedef __attribute__((ext_vector_type(8)))  float    v8f;
typedef __attribute__((ext_vector_type(4)))  int      v4i;

// Problem constants (match reference)
#define DQ  1024
#define TQ  64
#define HQ  16
#define DHQ 64

// ---------------------------------------------------------------------------
// CDNA5 async global->LDS path (ASYNCcnt-tracked), guarded so the file
// compiles on toolchains without the builtins.
// Builtin expects (v4i AS1*, v4i AS3*, imm offset, imm cpol).
// ---------------------------------------------------------------------------
#if __has_builtin(__builtin_amdgcn_global_load_async_to_lds_b128)
#define USE_ASYNC_LDS 1
#else
#define USE_ASYNC_LDS 0
#endif

#if USE_ASYNC_LDS
__device__ __forceinline__ void async_cp16B(const _Float16* g, _Float16* l) {
    // 16 bytes (8 halfs) global -> LDS, no VGPR round trip
    __builtin_amdgcn_global_load_async_to_lds_b128(
        (__attribute__((address_space(1))) v4i*)g,
        (__attribute__((address_space(3))) v4i*)l, 0, 0);
}
__device__ __forceinline__ void wait_async0() {
#if __has_builtin(__builtin_amdgcn_s_wait_asynccnt)
    __builtin_amdgcn_s_wait_asynccnt(0);
#else
    asm volatile("s_wait_asynccnt 0x0" ::: "memory");
#endif
}
#endif

// ---------------------------------------------------------------------------
// Fragment loaders per CDNA5 ISA 7.12.2 (wave32, 16-bit A 16x32 / B 32x16)
//   A: lanes 0-15  -> row M=lane,    halfs K {k0+0..7,  k0+16..23}
//      lanes 16-31 -> row M=lane-16, halfs K {k0+8..15, k0+24..31}
//   B: lanes 0-15  -> col N=lane,    halfs K {k0+0..7,  k0+16..23}
//      lanes 16-31 -> col N=lane-16, halfs K {k0+8..15, k0+24..31}
// ---------------------------------------------------------------------------
__device__ __forceinline__ v16h frag16(const _Float16* p0, const _Float16* p1) {
    v8h lo = *(const v8h*)p0;
    v8h hi = *(const v8h*)p1;
    v16h r;
#pragma unroll
    for (int i = 0; i < 8; ++i) { r[i] = lo[i]; r[i + 8] = hi[i]; }
    return r;
}

// ---------------------------------------------------------------------------
// f32 -> f16 elementwise convert (weights)
// ---------------------------------------------------------------------------
__global__ void __launch_bounds__(256)
f32_to_f16_kernel(const float* __restrict__ src, _Float16* __restrict__ dst, int n) {
    int i = (blockIdx.x * 256 + threadIdx.x) * 4;
    if (i + 3 < n) {
        float4 v = *(const float4*)(src + i);
        dst[i + 0] = (_Float16)v.x;
        dst[i + 1] = (_Float16)v.y;
        dst[i + 2] = (_Float16)v.z;
        dst[i + 3] = (_Float16)v.w;
    }
}

// ---------------------------------------------------------------------------
// LayerNorm over D=1024, output f16. One row per 256-thread block (8 wave32).
// ---------------------------------------------------------------------------
__global__ void __launch_bounds__(256)
ln_to_f16_kernel(const float* __restrict__ xin, const float* __restrict__ g,
                 const float* __restrict__ bb, _Float16* __restrict__ out) {
    const int row = blockIdx.x;
    const int tid = threadIdx.x;
    const float* xr = xin + (size_t)row * DQ;

    float xs[4];
    float s = 0.f, s2 = 0.f;
#pragma unroll
    for (int j = 0; j < 4; ++j) {
        xs[j] = xr[tid + j * 256];
        s += xs[j];
        s2 += xs[j] * xs[j];
    }
#pragma unroll
    for (int off = 16; off > 0; off >>= 1) {    // wave32 butterfly
        s  += __shfl_xor(s,  off, 32);
        s2 += __shfl_xor(s2, off, 32);
    }
    __shared__ float ws[8], ws2[8];
    __shared__ float mu_s, rstd_s;
    const int wv = tid >> 5;
    if ((tid & 31) == 0) { ws[wv] = s; ws2[wv] = s2; }
    __syncthreads();
    if (tid == 0) {
        float ts = 0.f, ts2 = 0.f;
#pragma unroll
        for (int i = 0; i < 8; ++i) { ts += ws[i]; ts2 += ws2[i]; }
        float mu  = ts / (float)DQ;
        float var = ts2 / (float)DQ - mu * mu;
        mu_s   = mu;
        rstd_s = rsqrtf(var + 1e-5f);
    }
    __syncthreads();
    const float mu = mu_s, rstd = rstd_s;
#pragma unroll
    for (int j = 0; j < 4; ++j) {
        int c = tid + j * 256;
        out[(size_t)row * DQ + c] = (_Float16)((xs[j] - mu) * rstd * g[c] + bb[c]);
    }
}

// ---------------------------------------------------------------------------
// WMMA GEMM: C[M,N] = A[M,K] @ W[N,K]^T + bias[N]
// Block tile 128x128, BK=32, 256 threads = 8 waves (2x4 wave grid),
// each wave: 64x32 = 4x2 tiles of v_wmma_f32_16x16x32_f16.
// Async path: double-buffered LDS stages filled by GLOBAL_LOAD_ASYNC_TO_LDS,
// one barrier per K-step, DMA overlaps the WMMA stream.
// ---------------------------------------------------------------------------
template <typename TOUT>
__global__ void __launch_bounds__(256)
gemm_bias_kernel(const _Float16* __restrict__ A, const _Float16* __restrict__ W,
                 const float* __restrict__ bias, TOUT* __restrict__ C,
                 int M, int N, int K) {
    constexpr int LDT = 40;                       // 32 + 8 halfs pad (16B-aligned rows)
#if USE_ASYNC_LDS
    constexpr int STAGES = 2;
#else
    constexpr int STAGES = 1;
#endif
    __shared__ _Float16 As[STAGES][128 * LDT];
    __shared__ _Float16 Bs[STAGES][128 * LDT];

    const int tid  = threadIdx.x;
    const int lane = tid & 31;
    const int wv   = tid >> 5;
    const int wm   = wv >> 2;                     // 0..1 (64 rows each)
    const int wn   = wv & 3;                      // 0..3 (32 cols each)
    const int m0   = blockIdx.x * 128;
    const int n0   = blockIdx.y * 128;

    v8f acc[4][2];
#pragma unroll
    for (int i = 0; i < 4; ++i)
#pragma unroll
        for (int j = 0; j < 2; ++j) acc[i][j] = {};

    const int lrow = tid >> 1;                    // 0..127
    const int lseg = (tid & 1) * 16;              // 0 or 16 halfs
    const _Float16* gA = A + (size_t)(m0 + lrow) * K + lseg;
    const _Float16* gW = W + (size_t)(n0 + lrow) * K + lseg;

#if USE_ASYNC_LDS
    // prime stage 0
    async_cp16B(gA,     &As[0][lrow * LDT + lseg]);
    async_cp16B(gA + 8, &As[0][lrow * LDT + lseg + 8]);
    async_cp16B(gW,     &Bs[0][lrow * LDT + lseg]);
    async_cp16B(gW + 8, &Bs[0][lrow * LDT + lseg + 8]);
#endif

    for (int k0 = 0; k0 < K; k0 += 32) {
#if USE_ASYNC_LDS
        const int cur = (k0 >> 5) & 1;
        wait_async0();            // my stage-`cur` DMA writes have landed in LDS
        __syncthreads();          // everyone's writes visible; prev compute done
        if (k0 + 32 < K) {        // kick DMA for the next stage; overlaps WMMAs
            const int nxt = cur ^ 1;
            async_cp16B(gA + k0 + 32,     &As[nxt][lrow * LDT + lseg]);
            async_cp16B(gA + k0 + 40,     &As[nxt][lrow * LDT + lseg + 8]);
            async_cp16B(gW + k0 + 32,     &Bs[nxt][lrow * LDT + lseg]);
            async_cp16B(gW + k0 + 40,     &Bs[nxt][lrow * LDT + lseg + 8]);
        }
#else
        constexpr int cur = 0;
        if (k0 + 32 < K) {        // -> global_prefetch_b8
            __builtin_prefetch(gA + k0 + 32, 0, 1);
            __builtin_prefetch(gW + k0 + 32, 0, 1);
        }
        *(v8h*)&As[0][lrow * LDT + lseg]     = *(const v8h*)(gA + k0);
        *(v8h*)&As[0][lrow * LDT + lseg + 8] = *(const v8h*)(gA + k0 + 8);
        *(v8h*)&Bs[0][lrow * LDT + lseg]     = *(const v8h*)(gW + k0);
        *(v8h*)&Bs[0][lrow * LDT + lseg + 8] = *(const v8h*)(gW + k0 + 8);
        __syncthreads();
#endif

        const int fr    = lane & 15;
        const int akoff = (lane >> 4) * 8;
        const int bkoff = (lane >> 4) * 8;
        v16h bfr[2];
#pragma unroll
        for (int j = 0; j < 2; ++j) {
            const _Float16* pb = &Bs[cur][(wn * 32 + j * 16 + fr) * LDT + bkoff];
            bfr[j] = frag16(pb, pb + 16);
        }
#pragma unroll
        for (int i = 0; i < 4; ++i) {
            const _Float16* pa = &As[cur][(wm * 64 + i * 16 + fr) * LDT + akoff];
            v16h afr = frag16(pa, pa + 16);
#pragma unroll
            for (int j = 0; j < 2; ++j)
                acc[i][j] = __builtin_amdgcn_wmma_f32_16x16x32_f16(
                    false, afr, false, bfr[j], (short)0, acc[i][j], false, false);
        }
#if !USE_ASYNC_LDS
        __syncthreads();
#endif
    }

    // Epilogue: C/D layout — VGPR r: (M = r + 8*(lane>=16), N = lane&15)
    const int crow = (lane >> 4) * 8;
    const int ccol = lane & 15;
#pragma unroll
    for (int i = 0; i < 4; ++i) {
#pragma unroll
        for (int j = 0; j < 2; ++j) {
            const int n  = n0 + wn * 32 + j * 16 + ccol;
            const float bv = bias[n];
#pragma unroll
            for (int r = 0; r < 8; ++r) {
                const int m = m0 + wm * 64 + i * 16 + r + crow;
                C[(size_t)m * N + n] = (TOUT)(acc[i][j][r] + bv);
            }
        }
    }
}

// ---------------------------------------------------------------------------
// Flash-style cross-attention, one block per (batch, head).
//   q: [T=64, D] f16, k/v: [B*L, D] f16, y: [B*T, D] f16 (unnormalized by 1/l
//   until the end). Online softmax over L in chunks of 128.
// ---------------------------------------------------------------------------
#define ACH  128          // L chunk
#define ALDQ 72           // qs  row stride (halfs)
#define ALDK 72           // ks  row stride
#define ALDV 136          // vt  row stride (transposed v: [d][l])
#define ALDP 136          // Pm  row stride
#define ALDS 132          // Sf  row stride (floats)

__global__ void __launch_bounds__(256)
attn_kernel(const _Float16* __restrict__ q, const _Float16* __restrict__ k,
            const _Float16* __restrict__ v, _Float16* __restrict__ y, int L) {
    extern __shared__ char smem[];
    _Float16* qs   = (_Float16*)smem;            // 64  x ALDQ
    _Float16* ks   = qs + 64 * ALDQ;             // 128 x ALDK
    _Float16* vt   = ks + 128 * ALDK;            // 64  x ALDV (v transposed)
    _Float16* Pm   = vt + 64 * ALDV;             // 64  x ALDP
    float* Sf      = (float*)(Pm + 64 * ALDP);   // 64  x ALDS
    float* redM    = Sf + 64 * ALDS;             // 64 x 4
    float* redS    = redM + 64 * 4;              // 64 x 4
    float* mArr    = redS + 64 * 4;              // 64
    float* lArr    = mArr + 64;                  // 64
    float* sclArr  = lArr + 64;                  // 64
    float* mNew    = sclArr + 64;                // 64

    const int tid  = threadIdx.x;
    const int lane = tid & 31;
    const int wv   = tid >> 5;
    const int b    = blockIdx.x >> 4;            // H = 16
    const int h    = blockIdx.x & 15;
    const float scale = 0.125f;                  // DH^-0.5

    // stage q head: 64 rows x 64 halfs
    {
        const int row = tid >> 2;
        const int seg = (tid & 3) * 16;
        const _Float16* gq = q + (size_t)row * DQ + h * DHQ + seg;
        *(v8h*)&qs[row * ALDQ + seg]     = *(const v8h*)gq;
        *(v8h*)&qs[row * ALDQ + seg + 8] = *(const v8h*)(gq + 8);
    }
    if (tid < 64) { mArr[tid] = -1e30f; lArr[tid] = 0.f; }

    // y accumulator tiles: wave wv owns rows ym*16..+16, cols ycb + {0,16}
    v8f yacc[2] = {{}, {}};
    const int ym  = wv >> 1;
    const int ycb = (wv & 1) * 32;
    __syncthreads();

    for (int l0 = 0; l0 < L; l0 += ACH) {
        const size_t rowbase = (size_t)b * L + l0;
        {   // k chunk: 128 rows x 64 halfs (async DMA where available)
            const int row = tid >> 1;
            const int seg = (tid & 1) * 32;
            const _Float16* gk = k + (rowbase + row) * DQ + h * DHQ + seg;
#pragma unroll
            for (int g = 0; g < 4; ++g) {
#if USE_ASYNC_LDS
                async_cp16B(gk + g * 8, &ks[row * ALDK + seg + g * 8]);
#else
                *(v8h*)&ks[row * ALDK + seg + g * 8] = *(const v8h*)(gk + g * 8);
#endif
            }
        }
        {   // v chunk transposed: vt[d][i]  (in-register transpose, sync loads)
            const int row   = tid >> 1;
            const int dbase = (tid & 1) * 32;
            const _Float16* gv = v + (rowbase + row) * DQ + h * DHQ + dbase;
#pragma unroll
            for (int g = 0; g < 4; ++g) {
                v8h vv = *(const v8h*)(gv + g * 8);
#pragma unroll
                for (int e = 0; e < 8; ++e)
                    vt[(dbase + g * 8 + e) * ALDV + row] = vv[e];
            }
        }
#if USE_ASYNC_LDS
        wait_async0();
#endif
        __syncthreads();

        {   // S = scale * q @ k^T  -> Sf[64][128]; wave: rows rm*16, cols cb + j*16
            const int rm    = wv >> 1;
            const int cb    = (wv & 1) * 64;
            const int fr    = lane & 15;
            const int akoff = (lane >> 4) * 8;
            const int bkoff = (lane >> 4) * 8;
#pragma unroll
            for (int j = 0; j < 4; ++j) {
                v8f s = {};
#pragma unroll
                for (int kk = 0; kk < DHQ; kk += 32) {
                    const _Float16* pa = &qs[(rm * 16 + fr) * ALDQ + kk + akoff];
                    const _Float16* pb = &ks[(cb + j * 16 + fr) * ALDK + kk + bkoff];
                    s = __builtin_amdgcn_wmma_f32_16x16x32_f16(
                        false, frag16(pa, pa + 16), false, frag16(pb, pb + 16),
                        (short)0, s, false, false);
                }
                const int t0 = rm * 16 + (lane >> 4) * 8;
                const int tc = cb + j * 16 + fr;
#pragma unroll
                for (int r = 0; r < 8; ++r)
                    Sf[(t0 + r) * ALDS + tc] = s[r] * scale;
            }
        }
        __syncthreads();

        {   // chunk row-max partials (4 threads per row)
            const int row = tid & 63, part = tid >> 6;
            const float* sp = &Sf[row * ALDS + part * 32];
            float mx = -1e30f;
#pragma unroll 8
            for (int c = 0; c < 32; ++c) mx = fmaxf(mx, sp[c]);
            redM[row * 4 + part] = mx;
        }
        __syncthreads();
        if (tid < 64) {
            float mc = fmaxf(fmaxf(redM[tid * 4], redM[tid * 4 + 1]),
                             fmaxf(redM[tid * 4 + 2], redM[tid * 4 + 3]));
            float mp = mArr[tid];
            float mn = fmaxf(mp, mc);
            mNew[tid]   = mn;
            sclArr[tid] = __expf(mp - mn);
        }
        __syncthreads();

        // rescale running y accumulators by exp(m_old - m_new)
#pragma unroll
        for (int j = 0; j < 2; ++j)
#pragma unroll
            for (int r = 0; r < 8; ++r)
                yacc[j][r] *= sclArr[ym * 16 + r + (lane >> 4) * 8];

        {   // P = exp(S - m_new) f16, plus row-sum partials
            const int row = tid & 63, part = tid >> 6;
            const float mn = mNew[row];
            const float* sp = &Sf[row * ALDS + part * 32];
            _Float16* pp = &Pm[row * ALDP + part * 32];
            float sm = 0.f;
#pragma unroll 8
            for (int c = 0; c < 32; ++c) {
                float pv = __expf(sp[c] - mn);
                pp[c] = (_Float16)pv;
                sm += pv;
            }
            redS[row * 4 + part] = sm;
        }
        __syncthreads();
        if (tid < 64) {
            lArr[tid] = lArr[tid] * sclArr[tid] +
                        (redS[tid * 4] + redS[tid * 4 + 1] +
                         redS[tid * 4 + 2] + redS[tid * 4 + 3]);
            mArr[tid] = mNew[tid];
        }

        {   // y += P @ v_chunk   (B-fragments read K-contiguous from vt)
            const int fr    = lane & 15;
            const int akoff = (lane >> 4) * 8;
            const int bkoff = (lane >> 4) * 8;
#pragma unroll
            for (int kk = 0; kk < ACH; kk += 32) {
                const _Float16* pa = &Pm[(ym * 16 + fr) * ALDP + kk + akoff];
                v16h af = frag16(pa, pa + 16);
#pragma unroll
                for (int j = 0; j < 2; ++j) {
                    const _Float16* pb = &vt[(ycb + j * 16 + fr) * ALDV + kk + bkoff];
                    yacc[j] = __builtin_amdgcn_wmma_f32_16x16x32_f16(
                        false, af, false, frag16(pb, pb + 16),
                        (short)0, yacc[j], false, false);
                }
            }
        }
        __syncthreads();
    }

    // write y / l  (f16) to y[b*T + t][h*64 + d]
    {
        const int fr = lane & 15;
#pragma unroll
        for (int j = 0; j < 2; ++j)
#pragma unroll
            for (int r = 0; r < 8; ++r) {
                const int t = ym * 16 + r + (lane >> 4) * 8;
                const int d = ycb + j * 16 + fr;
                y[((size_t)b * TQ + t) * DQ + h * DHQ + d] =
                    (_Float16)(yacc[j][r] / lArr[t]);
            }
    }
}

// ---------------------------------------------------------------------------
// Host launcher
// ---------------------------------------------------------------------------
extern "C" void kernel_launch(void* const* d_in, const int* in_sizes, int n_in,
                              void* d_out, int out_size, void* d_ws, size_t ws_size,
                              hipStream_t stream) {
    const float* x      = (const float*)d_in[0];
    const float* qry    = (const float*)d_in[1];
    const float* lnq_g  = (const float*)d_in[2];
    const float* lnq_b  = (const float*)d_in[3];
    const float* lnkv_g = (const float*)d_in[4];
    const float* lnkv_b = (const float*)d_in[5];
    const float* k_w    = (const float*)d_in[6];
    const float* k_b    = (const float*)d_in[7];
    const float* v_w    = (const float*)d_in[8];
    const float* v_b    = (const float*)d_in[9];
    const float* out_w  = (const float*)d_in[10];
    const float* out_b  = (const float*)d_in[11];
    float* outp = (float*)d_out;

    const int BL = in_sizes[0] / DQ;             // 65536
    const int Bq = BL / 4096;                    // 16
    const int Lq = 4096;

    // workspace carve-up
    char* ws = (char*)d_ws;
    _Float16* kw16  = (_Float16*)ws; ws += (size_t)DQ * DQ * 2;
    _Float16* vw16  = (_Float16*)ws; ws += (size_t)DQ * DQ * 2;
    _Float16* ow16  = (_Float16*)ws; ws += (size_t)DQ * DQ * 2;
    _Float16* q16   = (_Float16*)ws; ws += (size_t)TQ * DQ * 2;
    _Float16* xkv16 = (_Float16*)ws; ws += (size_t)BL * DQ * 2;
    _Float16* k16   = (_Float16*)ws; ws += (size_t)BL * DQ * 2;
    _Float16* v16   = (_Float16*)ws; ws += (size_t)BL * DQ * 2;
    _Float16* y16   = (_Float16*)ws; ws += (size_t)Bq * TQ * DQ * 2;

    // 1) weights -> f16
    const int nW = DQ * DQ;
    f32_to_f16_kernel<<<nW / 1024, 256, 0, stream>>>(k_w, kw16, nW);
    f32_to_f16_kernel<<<nW / 1024, 256, 0, stream>>>(v_w, vw16, nW);
    f32_to_f16_kernel<<<nW / 1024, 256, 0, stream>>>(out_w, ow16, nW);

    // 2) LayerNorms -> f16
    ln_to_f16_kernel<<<TQ, 256, 0, stream>>>(qry, lnq_g, lnq_b, q16);
    ln_to_f16_kernel<<<BL, 256, 0, stream>>>(x, lnkv_g, lnkv_b, xkv16);

    // 3) K / V projections (WMMA GEMM, async double-buffered staging)
    dim3 gkv(BL / 128, DQ / 128);
    gemm_bias_kernel<_Float16><<<gkv, 256, 0, stream>>>(xkv16, kw16, k_b, k16, BL, DQ, DQ);
    gemm_bias_kernel<_Float16><<<gkv, 256, 0, stream>>>(xkv16, vw16, v_b, v16, BL, DQ, DQ);

    // 4) attention (dynamic LDS ~99 KB; CDNA5 WGP has 320 KB)
    const size_t attn_lds =
        (size_t)(64 * ALDQ + 128 * ALDK + 64 * ALDV + 64 * ALDP) * 2 +
        (size_t)(64 * ALDS + 64 * 4 + 64 * 4 + 4 * 64) * 4;
    (void)hipFuncSetAttribute((const void*)attn_kernel,
                              hipFuncAttributeMaxDynamicSharedMemorySize, (int)attn_lds);
    attn_kernel<<<Bq * HQ, 256, attn_lds, stream>>>(q16, k16, v16, y16, Lq);

    // 5) output projection -> f32 d_out
    dim3 go((Bq * TQ) / 128, DQ / 128);
    gemm_bias_kernel<float><<<go, 256, 0, stream>>>(y16, ow16, out_b, outp, Bq * TQ, DQ, DQ);
}
